// URAClusteringGNN_81466939670634
// MI455X (gfx1250) — matrix operand order repaired and checked by reading, not compiled, not run on hardware
//
#include <hip/hip_runtime.h>
#include <math.h>

#define NODES  20000
#define EDGES  640000
#define IN_DIM 2
#define HEADS  4
#define CPH    64
#define HCSZ   (HEADS * CPH)   // 256
#define EDIM   3
#define NCLUST 4
#define SEPS   1e-16f

typedef __attribute__((ext_vector_type(16))) _Float16 v16h;
typedef __attribute__((ext_vector_type(8)))  _Float16 v8h;
typedef __attribute__((ext_vector_type(8)))  float    v8f;

__device__ __forceinline__ float elu1(float x) {
    return x > 0.f ? x : (__expf(x) - 1.f);
}

__device__ __forceinline__ void atomicMaxFloat(float* addr, float val) {
    if (val >= 0.f) {
        atomicMax((int*)addr, __float_as_int(val));
    } else {
        atomicMin((unsigned int*)addr, __float_as_uint(val));
    }
}

// ---------------------------------------------------------------------------
// generic fill
__global__ void fill_f32(float* p, float v, int n) {
    int tid = blockIdx.x * blockDim.x + threadIdx.x;
    if (tid < n) p[tid] = v;
}

// input projection: h = elu(x @ Wp + bp)   [N,2] @ [2,64]
__global__ void proj_kernel(const float* __restrict__ x, const float* __restrict__ Wp,
                            const float* __restrict__ bp, float* __restrict__ h) {
    int tid = blockIdx.x * blockDim.x + threadIdx.x;
    if (tid >= NODES * CPH) return;
    int n = tid / CPH, c = tid - n * CPH;
    float v = bp[c] + x[n * 2 + 0] * Wp[0 * CPH + c] + x[n * 2 + 1] * Wp[1 * CPH + c];
    h[tid] = elu1(v);
}

// f32 -> f16 activation conversion
__global__ void cvt_f32_to_f16(const float* __restrict__ src, _Float16* __restrict__ dst, int n) {
    int tid = blockIdx.x * blockDim.x + threadIdx.x;
    if (tid < n) dst[tid] = (_Float16)src[tid];
}

// W [Kin][Nout] f32 -> Wt [Nout][Kin] f16 (transposed for B-fragment loads)
__global__ void cvt_w_t_f16(const float* __restrict__ W, _Float16* __restrict__ Wt,
                            int Kin, int Nout) {
    int tid = blockIdx.x * blockDim.x + threadIdx.x;
    if (tid >= Kin * Nout) return;
    int nn = tid % Nout, k = tid / Nout;
    Wt[(size_t)nn * Kin + k] = (_Float16)W[(size_t)k * Nout + nn];
}

// ---------------------------------------------------------------------------
// WMMA GEMM: D[M,N] = A[M,K](f16) * Bt[N,K](f16, B transposed), f32 accumulate.
// One wave32 per 16x16 output tile, K stepped by 32 via v_wmma_f32_16x16x32_f16.
__global__ void wmma_gemm_f16f32(const _Float16* __restrict__ A,
                                 const _Float16* __restrict__ Bt,
                                 float* __restrict__ D,
                                 int M, int K, int Nn) {
    int wavesPerBlk = blockDim.x >> 5;
    int wave  = blockIdx.x * wavesPerBlk + (threadIdx.x >> 5);
    int tilesN = Nn >> 4;
    int tM = wave / tilesN;
    int tN = wave - tM * tilesN;
    if (tM * 16 >= M) return;

    int lane = threadIdx.x & 31;
    int hi   = lane >> 4;      // half-wave select
    int r    = lane & 15;

    const _Float16* Ar = A  + (size_t)(tM * 16 + r) * K;   // row M = r
    const _Float16* Br = Bt + (size_t)(tN * 16 + r) * K;   // col N = r

    v8f acc = {};
    for (int k0 = 0; k0 < K; k0 += 32) {
        // A: lane holds K = k0 + {j + 8*(j>=8) + 8*hi}  -> two contiguous 8-half runs
        v8h a0 = *(const v8h*)(Ar + k0 + 8 * hi);
        v8h a1 = *(const v8h*)(Ar + k0 + 16 + 8 * hi);
        // B: lane holds K = k0 + {j + 8*(j>=8) + 8*hi}  -> two contiguous 8-half runs
        v8h b0 = *(const v8h*)(Br + k0 + 8 * hi);
        v8h b1 = *(const v8h*)(Br + k0 + 16 + 8 * hi);
        v16h a, b;
#pragma unroll
        for (int i = 0; i < 8; ++i) {
            a[i] = a0[i]; a[i + 8] = a1[i];
            b[i] = b0[i]; b[i + 8] = b1[i];
        }
        acc = __builtin_amdgcn_wmma_f32_16x16x32_f16(
            false, a, false, b, (short)0, acc, false, false);
    }
    // D layout: VGPR v -> row v + 8*hi, col r
    float* Dp = D + (size_t)(tM * 16 + 8 * hi) * Nn + tN * 16 + r;
#pragma unroll
    for (int v = 0; v < 8; ++v) Dp[(size_t)v * Nn] = acc[v];
}

// ---------------------------------------------------------------------------
// per-(node,head) attention logits: asrc = <xh, a_src>, adst = <xh, a_dst>
__global__ void alpha_node_kernel(const float* __restrict__ xh,
                                  const float* __restrict__ as_, const float* __restrict__ ad_,
                                  float* __restrict__ asrc, float* __restrict__ adst,
                                  int heads) {
    int tid = blockIdx.x * blockDim.x + threadIdx.x;
    if (tid >= NODES * heads) return;
    int n = tid / heads, hh = tid - n * heads;
    const float* xp = xh + (size_t)n * heads * CPH + hh * CPH;
    const float* ap = as_ + hh * CPH;
    const float* dp = ad_ + hh * CPH;
    float sa = 0.f, sd = 0.f;
#pragma unroll 8
    for (int c = 0; c < CPH; ++c) { float xv = xp[c]; sa += xv * ap[c]; sd += xv * dp[c]; }
    asrc[tid] = sa; adst[tid] = sd;
}

// ve[d][h] = sum_c We[d][h*C+c] * ae[h][c]   (3 x heads, collapses the edge einsum)
__global__ void ve_kernel(const float* __restrict__ We, const float* __restrict__ ae,
                          float* __restrict__ ve, int heads) {
    int tid = threadIdx.x;
    if (tid >= EDIM * heads) return;
    int d = tid / heads, hh = tid - d * heads;
    float s = 0.f;
    for (int c = 0; c < CPH; ++c)
        s += We[(size_t)d * heads * CPH + hh * CPH + c] * ae[hh * CPH + c];
    ve[d * heads + hh] = s;
}

// per-edge attention logit + leaky_relu + segment max over dst
__global__ void edge_alpha_kernel(const int* __restrict__ ei, const float* __restrict__ eattr,
                                  const float* __restrict__ asrc, const float* __restrict__ adst,
                                  const float* __restrict__ ve, float* __restrict__ alpha,
                                  float* __restrict__ m, int heads) {
    int tid = blockIdx.x * blockDim.x + threadIdx.x;
    if (tid >= EDGES * heads) return;
    int e = tid / heads, hh = tid - e * heads;
    int s = ei[e], d = ei[EDGES + e];
    float av = asrc[s * heads + hh] + adst[d * heads + hh];
#pragma unroll
    for (int k = 0; k < EDIM; ++k) av += eattr[e * EDIM + k] * ve[k * heads + hh];
    av = av > 0.f ? av : 0.2f * av;               // leaky_relu(0.2)
    alpha[tid] = av;
    atomicMaxFloat(&m[d * heads + hh], av);
}

// ea = exp(alpha - m[dst]); segment sum s[dst] += ea  (alpha updated in place)
__global__ void edge_exp_kernel(const int* __restrict__ ei, float* __restrict__ alpha,
                                const float* __restrict__ m, float* __restrict__ sbuf,
                                int heads) {
    int tid = blockIdx.x * blockDim.x + threadIdx.x;
    if (tid >= EDGES * heads) return;
    int e = tid / heads, hh = tid - e * heads;
    int d = ei[EDGES + e];
    float v = __expf(alpha[tid] - m[d * heads + hh]);
    alpha[tid] = v;
    atomicAdd(&sbuf[d * heads + hh], v);
}

// message aggregation: acc[dst] += xh[src] * ea/(s[dst]+eps)
__global__ void edge_msg_kernel(const int* __restrict__ ei, const float* __restrict__ xh,
                                const float* __restrict__ ea, const float* __restrict__ sbuf,
                                float* __restrict__ accp, int heads) {
    long long tid = (long long)blockIdx.x * blockDim.x + threadIdx.x;
    const int F = heads * CPH;
    if (tid >= (long long)EDGES * F) return;
    int e   = (int)(tid / F);
    int rem = (int)(tid - (long long)e * F);
    int hh  = rem >> 6;                            // CPH == 64
    int s = ei[e], d = ei[EDGES + e];
    float w = ea[e * heads + hh] / (sbuf[d * heads + hh] + SEPS);
    atomicAdd(&accp[(size_t)d * F + rem], xh[(size_t)s * F + rem] * w);
}

// out = (acc + bias), optional ELU
__global__ void finalize_kernel(const float* __restrict__ accp, const float* __restrict__ b,
                                float* __restrict__ hout, int nout, int do_elu) {
    int tid = blockIdx.x * blockDim.x + threadIdx.x;
    if (tid >= NODES * nout) return;
    float v = accp[tid] + b[tid % nout];
    hout[tid] = do_elu ? elu1(v) : v;
}

// classifier + softmax over 4 clusters
__global__ void classifier_kernel(const float* __restrict__ h, const float* __restrict__ Wc,
                                  const float* __restrict__ bc, float* __restrict__ out) {
    int n = blockIdx.x * blockDim.x + threadIdx.x;
    if (n >= NODES) return;
    float logit[NCLUST];
#pragma unroll
    for (int k = 0; k < NCLUST; ++k) logit[k] = bc[k];
    const float* hp = h + (size_t)n * CPH;
    for (int c = 0; c < CPH; ++c) {
        float hv = hp[c];
#pragma unroll
        for (int k = 0; k < NCLUST; ++k) logit[k] += hv * Wc[c * NCLUST + k];
    }
    float mx = logit[0];
#pragma unroll
    for (int k = 1; k < NCLUST; ++k) mx = fmaxf(mx, logit[k]);
    float sum = 0.f;
#pragma unroll
    for (int k = 0; k < NCLUST; ++k) { logit[k] = __expf(logit[k] - mx); sum += logit[k]; }
    float inv = 1.f / sum;
#pragma unroll
    for (int k = 0; k < NCLUST; ++k) out[n * NCLUST + k] = logit[k] * inv;
}

// ---------------------------------------------------------------------------
static inline int cdiv(long long a, long long b) { return (int)((a + b - 1) / b); }

extern "C" void kernel_launch(void* const* d_in, const int* in_sizes, int n_in,
                              void* d_out, int out_size, void* d_ws, size_t ws_size,
                              hipStream_t stream) {
    (void)in_sizes; (void)n_in; (void)out_size; (void)ws_size;

    const float* x      = (const float*)d_in[0];
    const int*   ei     = (const int*)  d_in[1];
    const float* eattr  = (const float*)d_in[2];
    const float* Wp     = (const float*)d_in[3];
    const float* bp     = (const float*)d_in[4];
    const float* W1  = (const float*)d_in[5];  const float* We1 = (const float*)d_in[6];
    const float* as1 = (const float*)d_in[7];  const float* ad1 = (const float*)d_in[8];
    const float* ae1 = (const float*)d_in[9];  const float* b1  = (const float*)d_in[10];
    const float* W2  = (const float*)d_in[11]; const float* We2 = (const float*)d_in[12];
    const float* as2 = (const float*)d_in[13]; const float* ad2 = (const float*)d_in[14];
    const float* ae2 = (const float*)d_in[15]; const float* b2  = (const float*)d_in[16];
    const float* W3  = (const float*)d_in[17]; const float* We3 = (const float*)d_in[18];
    const float* as3 = (const float*)d_in[19]; const float* ad3 = (const float*)d_in[20];
    const float* ae3 = (const float*)d_in[21]; const float* b3  = (const float*)d_in[22];
    const float* Wc  = (const float*)d_in[23]; const float* bc  = (const float*)d_in[24];
    float* out = (float*)d_out;

    // workspace carve-up (256B aligned)
    char* wp_ = (char*)d_ws;
    auto alloc = [&](size_t bytes) { char* p = wp_; wp_ += (bytes + 255) & ~(size_t)255; return p; };
    float*    hcur  = (float*)   alloc((size_t)NODES * HCSZ * 4);
    float*    xh    = (float*)   alloc((size_t)NODES * HCSZ * 4);
    float*    acc   = (float*)   alloc((size_t)NODES * HCSZ * 4);
    _Float16* a16   = (_Float16*)alloc((size_t)NODES * HCSZ * 2);
    _Float16* wt16  = (_Float16*)alloc((size_t)HCSZ * HCSZ * 2);
    float*    asrc  = (float*)   alloc((size_t)NODES * HEADS * 4);
    float*    adst  = (float*)   alloc((size_t)NODES * HEADS * 4);
    float*    mbuf  = (float*)   alloc((size_t)NODES * HEADS * 4);
    float*    sbuf  = (float*)   alloc((size_t)NODES * HEADS * 4);
    float*    albuf = (float*)   alloc((size_t)EDGES * HEADS * 4);
    float*    vebuf = (float*)   alloc(256);

    const int T = 256;

    // input projection
    proj_kernel<<<cdiv((long long)NODES * CPH, T), T, 0, stream>>>(x, Wp, bp, hcur);

    auto run_gat = [&](int Kin, int heads,
                       const float* W, const float* We, const float* as_,
                       const float* ad_, const float* ae_, const float* b_,
                       int do_elu) {
        const int nout = heads * CPH;
        // f16 staging
        cvt_f32_to_f16<<<cdiv((long long)NODES * Kin, T), T, 0, stream>>>(hcur, a16, NODES * Kin);
        cvt_w_t_f16<<<cdiv((long long)Kin * nout, T), T, 0, stream>>>(W, wt16, Kin, nout);
        // node GEMM via WMMA: one wave per 16x16 tile, 4 waves / block
        int tiles = (NODES / 16) * (nout / 16);
        wmma_gemm_f16f32<<<cdiv(tiles, 4), 128, 0, stream>>>(a16, wt16, xh, NODES, Kin, nout);
        // attention scalars
        alpha_node_kernel<<<cdiv((long long)NODES * heads, T), T, 0, stream>>>(
            xh, as_, ad_, asrc, adst, heads);
        ve_kernel<<<1, 32, 0, stream>>>(We, ae_, vebuf, heads);
        // init segment buffers
        fill_f32<<<cdiv((long long)NODES * heads, T), T, 0, stream>>>(mbuf, -3.0e38f, NODES * heads);
        fill_f32<<<cdiv((long long)NODES * heads, T), T, 0, stream>>>(sbuf, 0.f, NODES * heads);
        fill_f32<<<cdiv((long long)NODES * nout, T), T, 0, stream>>>(acc, 0.f, NODES * nout);
        // edge pipeline: logits -> segment softmax -> weighted aggregation
        edge_alpha_kernel<<<cdiv((long long)EDGES * heads, T), T, 0, stream>>>(
            ei, eattr, asrc, adst, vebuf, albuf, mbuf, heads);
        edge_exp_kernel<<<cdiv((long long)EDGES * heads, T), T, 0, stream>>>(
            ei, albuf, mbuf, sbuf, heads);
        edge_msg_kernel<<<cdiv((long long)EDGES * heads * CPH, T), T, 0, stream>>>(
            ei, xh, albuf, sbuf, acc, heads);
        // bias + activation, written back into hcur (in place is safe: last reader done)
        finalize_kernel<<<cdiv((long long)NODES * nout, T), T, 0, stream>>>(
            acc, b_, hcur, nout, do_elu);
    };

    run_gat(CPH,  HEADS, W1, We1, as1, ad1, ae1, b1, 1);   // conv1: 64 -> 256, ELU
    run_gat(HCSZ, HEADS, W2, We2, as2, ad2, ae2, b2, 1);   // conv2: 256 -> 256, ELU
    run_gat(HCSZ, 1,     W3, We3, as3, ad3, ae3, b3, 0);   // conv3: 256 -> 64, no ELU

    classifier_kernel<<<cdiv(NODES, T), T, 0, stream>>>(hcur, Wc, bc, out);
}
